// SlotAttention_43868795961955
// MI455X (gfx1250) — matrix operand order repaired and checked by reading, not compiled, not run on hardware
//
#include <hip/hip_runtime.h>
#include <hip/hip_bf16.h>

// SlotAttention for MI455X (gfx1250): bf16 WMMA everywhere, f32 accumulate.
// B=64, N=4096, D=256, S=8, ITERS=3.

#define B_    64
#define N_    4096
#define D_    256
#define S_    8
#define ITERS_ 3
#define EPS_  1e-5f

typedef __bf16 bf16_t;
typedef bf16_t  bf16x16  __attribute__((ext_vector_type(16)));
typedef float   floatx8  __attribute__((ext_vector_type(8)));
typedef unsigned short ushort8_t  __attribute__((ext_vector_type(8)));
typedef unsigned short ushort16_t __attribute__((ext_vector_type(16)));

// ---------------- helpers ----------------

__device__ __forceinline__ unsigned short f2bf(float f) {
  unsigned int u = __builtin_bit_cast(unsigned int, f);
  u += 0x7FFFu + ((u >> 16) & 1u);           // round-to-nearest-even
  return (unsigned short)(u >> 16);
}

__device__ __forceinline__ floatx8 zero8() {
  floatx8 c;
#pragma unroll
  for (int i = 0; i < 8; ++i) c[i] = 0.f;
  return c;
}

// A-fragment (16x32 bf16) from row-major storage, row stride ld (elements).
// ISA layout: lane<16 holds row=lane, K = {ks+0..7, ks+16..23};
//             lane>=16 holds row=lane-16, K = {ks+8..15, ks+24..31}.
template <bool PAD>
__device__ __forceinline__ bf16x16 load_a(const unsigned short* base, int ld,
                                          int lane, int valid_rows) {
  ushort16_t r;
  int row = lane & 15;
  if (PAD && row >= valid_rows) {
#pragma unroll
    for (int i = 0; i < 16; ++i) r[i] = 0;
  } else {
    const unsigned short* p = base + row * ld + ((lane & 16) ? 8 : 0);
    ushort8_t lo = *(const ushort8_t*)(p);
    ushort8_t hi = *(const ushort8_t*)(p + 16);
#pragma unroll
    for (int i = 0; i < 8; ++i) { r[i] = lo[i]; r[i + 8] = hi[i]; }
  }
  return __builtin_bit_cast(bf16x16, r);
}

// B-fragment (32x16 bf16) where "base" is row-major with row = output column
// (i.e. W[e][d] for X@W^T GEMMs), row stride ld.
// ISA layout: lane<16 -> col=lane, K={ks+0..7, ks+16..23}; lane>=16 -> col=lane-16, K={ks+8..15, ks+24..31}.
__device__ __forceinline__ bf16x16 load_b(const unsigned short* base, int ld,
                                          int lane) {
  const unsigned short* p = base + (lane & 15) * ld + ((lane & 16) ? 8 : 0);
  ushort8_t lo = *(const ushort8_t*)(p);
  ushort8_t hi = *(const ushort8_t*)(p + 16);
  ushort16_t r;
#pragma unroll
  for (int i = 0; i < 8; ++i) { r[i] = lo[i]; r[i + 8] = hi[i]; }
  return __builtin_bit_cast(bf16x16, r);
}

__device__ __forceinline__ floatx8 wmma_bf(bf16x16 a, bf16x16 b, floatx8 c) {
  return __builtin_amdgcn_wmma_f32_16x16x32_bf16(false, a, false, b, (short)0,
                                                 c, false, false);
}

__device__ __forceinline__ float wsum(float v) {
#pragma unroll
  for (int o = 16; o > 0; o >>= 1) v += __shfl_xor(v, o, 32);
  return v;
}
__device__ __forceinline__ float wmax(float v) {
#pragma unroll
  for (int o = 16; o > 0; o >>= 1) v = fmaxf(v, __shfl_xor(v, o, 32));
  return v;
}

// ---------------- kernels ----------------

__global__ void cast_bf16_kernel(const float* __restrict__ src,
                                 unsigned short* __restrict__ dst, int n) {
  int i = blockIdx.x * 256 + threadIdx.x;
  if (i < n) dst[i] = f2bf(src[i]);
}

__global__ void init_slots_kernel(const float* __restrict__ noise,
                                  const float* __restrict__ mu,
                                  const float* __restrict__ sigma,
                                  float* __restrict__ slots) {
  int i = blockIdx.x * 256 + threadIdx.x;   // over B*S*D
  int d = i & (D_ - 1);
  slots[i] = mu[d] + __expf(sigma[d]) * noise[i];
}

// LayerNorm over x rows -> bf16. One wave per row, 8 rows per block.
__global__ void ln_x_kernel(const float* __restrict__ x,
                            const float* __restrict__ g,
                            const float* __restrict__ be,
                            unsigned short* __restrict__ xn) {
  int w = threadIdx.x >> 5, lane = threadIdx.x & 31;
  long row = (long)blockIdx.x * 8 + w;
  const float* xr = x + row * D_;
  float v[8];
  float s = 0.f;
#pragma unroll
  for (int j = 0; j < 8; ++j) { v[j] = xr[lane + 32 * j]; s += v[j]; }
  s = wsum(s);
  float mean = s * (1.f / D_);
  float q = 0.f;
#pragma unroll
  for (int j = 0; j < 8; ++j) { float d = v[j] - mean; q += d * d; }
  q = wsum(q);
  float rstd = rsqrtf(q * (1.f / D_) + EPS_);
  unsigned short* orow = xn + row * D_;
#pragma unroll
  for (int j = 0; j < 8; ++j) {
    int idx = lane + 32 * j;
    orow[idx] = f2bf((v[j] - mean) * rstd * g[idx] + be[idx]);
  }
}

// k = xn@Wk^T + bk (row-major bf16); v = xn@Wv^T + bv stored transposed per
// batch: vT[b][e][n] bf16. Block = 32 rows x 256 cols, 8 waves (2x4 tiling),
// each wave owns a 16x64 tile for BOTH outputs (shared A fragment).
// B-fragments for a K-step are preloaded into independent registers so the
// scheduler can keep 8 WMMAs in flight behind partial s_wait_loadcnt.
__global__ void kv_kernel(const unsigned short* __restrict__ xn,
                          const unsigned short* __restrict__ Wk,
                          const float* __restrict__ bk,
                          const unsigned short* __restrict__ Wv,
                          const float* __restrict__ bv,
                          unsigned short* __restrict__ kout,
                          unsigned short* __restrict__ vT) {
  int wv = threadIdx.x >> 5, lane = threadIdx.x & 31;
  int rw = wv >> 2, cw = wv & 3;
  long rowbase = (long)blockIdx.x * 32 + rw * 16;
  int colbase = cw * 64;
  floatx8 ck[4], cv[4];
#pragma unroll
  for (int t = 0; t < 4; ++t) { ck[t] = zero8(); cv[t] = zero8(); }
  const unsigned short* arow = xn + rowbase * D_;
#pragma unroll
  for (int kk = 0; kk < 8; ++kk) {
    int ks = kk * 32;
    bf16x16 a = load_a<false>(arow + ks, D_, lane, 16);
    bf16x16 bkf[4], bvf[4];
#pragma unroll
    for (int t = 0; t < 4; ++t) {
      bkf[t] = load_b(Wk + (colbase + t * 16) * D_ + ks, D_, lane);
      bvf[t] = load_b(Wv + (colbase + t * 16) * D_ + ks, D_, lane);
    }
#pragma unroll
    for (int t = 0; t < 4; ++t) {
      ck[t] = wmma_bf(a, bkf[t], ck[t]);
      cv[t] = wmma_bf(a, bvf[t], cv[t]);
    }
  }
  int rr = (lane & 16) ? 8 : 0;
  long bidx = rowbase >> 12;             // batch (N_=4096 rows per batch)
  int nloc = (int)(rowbase & (N_ - 1));
#pragma unroll
  for (int t = 0; t < 4; ++t) {
    int e = colbase + t * 16 + (lane & 15);
    float biask = bk[e], biasv = bv[e];
    unsigned short* kp = kout + (rowbase + rr) * D_ + e;
#pragma unroll
    for (int r = 0; r < 8; ++r) kp[r * D_] = f2bf(ck[t][r] + biask);
    ushort8_t pk;
#pragma unroll
    for (int r = 0; r < 8; ++r) pk[r] = f2bf(cv[t][r] + biasv);
    *(ushort8_t*)(vT + (bidx * D_ + e) * (long)N_ + nloc + rr) = pk;
  }
}

// Per-batch attention: LN(slots), q proj, logits = q@k^T, softmax over N in
// LDS, updates = attn@v. One block (8 waves) per batch; logits (8x4096 f32)
// and probs (8x4096 bf16) live in LDS (~200KB of the 320KB WGP pool).
__global__ void attn_kernel(const float* __restrict__ slots,
                            const unsigned short* __restrict__ Wq,
                            const float* __restrict__ bq,
                            const unsigned short* __restrict__ kbf,
                            const unsigned short* __restrict__ vT,
                            const float* __restrict__ g_sl,
                            const float* __restrict__ be_sl,
                            float* __restrict__ updates) {
  extern __shared__ char smem[];
  float* logits = (float*)smem;                             // 8*4096 f32
  unsigned short* attnp = (unsigned short*)(logits + 8 * N_); // 8*4096 bf16
  unsigned short* snb = attnp + 8 * N_;                     // 8*256 bf16
  unsigned short* qb = snb + 8 * D_;                        // 8*256 bf16

  int b = blockIdx.x;
  int w = threadIdx.x >> 5, lane = threadIdx.x & 31;

  // 1. LN over the 8 slots of this batch (wave w -> row w)
  if (w < S_) {
    const float* sr = slots + ((long)b * S_ + w) * D_;
    float v[8];
    float s = 0.f;
#pragma unroll
    for (int j = 0; j < 8; ++j) { v[j] = sr[lane + 32 * j]; s += v[j]; }
    s = wsum(s);
    float mean = s * (1.f / D_);
    float q = 0.f;
#pragma unroll
    for (int j = 0; j < 8; ++j) { float d = v[j] - mean; q += d * d; }
    q = wsum(q);
    float rstd = rsqrtf(q * (1.f / D_) + EPS_);
#pragma unroll
    for (int j = 0; j < 8; ++j) {
      int idx = lane + 32 * j;
      snb[w * D_ + idx] = f2bf((v[j] - mean) * rstd * g_sl[idx] + be_sl[idx]);
    }
  }
  __syncthreads();

  // 2. q = sn @ Wq^T + bq  (M=16 padded, N=256, K=256); 2 col tiles per wave.
  // A-fragments depend only on kk -> hoist once, reuse for both tiles.
  {
    bf16x16 aq[8];
#pragma unroll
    for (int kk = 0; kk < 8; ++kk)
      aq[kk] = load_a<true>(snb + kk * 32, D_, lane, S_);
    for (int tt = 0; tt < 2; ++tt) {
      int e0 = w * 32 + tt * 16;
      floatx8 c = zero8();
#pragma unroll
      for (int kk = 0; kk < 8; ++kk)
        c = wmma_bf(aq[kk], load_b(Wq + e0 * D_ + kk * 32, D_, lane), c);
      if (lane < 16) {
        int e = e0 + lane;
        float bias = bq[e];
#pragma unroll
        for (int r = 0; r < 8; ++r) qb[r * D_ + e] = f2bf(c[r] + bias);
      }
    }
  }
  __syncthreads();

  // 3. logits = q @ k^T  (k row-major bf16 is the B operand).
  // A-fragments (from qb) are tile-invariant -> hoist across all 32 tiles.
  {
    bf16x16 aq[8];
#pragma unroll
    for (int kk = 0; kk < 8; ++kk)
      aq[kk] = load_a<true>(qb + kk * 32, D_, lane, S_);
    for (int t = w; t < N_ / 16; t += 8) {
      int n0 = t * 16;
      const unsigned short* kb = kbf + ((long)b * N_ + n0) * D_;
      // prefetch this wave's next k tile (8 tiles ahead = 8KB, 256B/lane)
      __builtin_prefetch(kb + 128 * D_ + (lane << 7), 0, 1);
      floatx8 c = zero8();
#pragma unroll
      for (int kk = 0; kk < 8; ++kk)
        c = wmma_bf(aq[kk], load_b(kb + kk * 32, D_, lane), c);
      if (lane < 16) {
        int n = n0 + lane;
#pragma unroll
        for (int r = 0; r < 8; ++r) logits[r * N_ + n] = c[r];
      }
    }
  }
  __syncthreads();

  // 4. softmax over N per row (wave w -> row w)
  if (w < S_) {
    float* lr = logits + w * N_;
    float mx = -3.0e38f;
    for (int j = lane; j < N_; j += 32) mx = fmaxf(mx, lr[j]);
    mx = wmax(mx);
    float s = 0.f;
    for (int j = lane; j < N_; j += 32) {
      float e = __expf(lr[j] - mx);
      lr[j] = e;
      s += e;
    }
    s = wsum(s);
    float inv = 1.f / s;
    for (int j = lane; j < N_; j += 32) attnp[w * N_ + j] = f2bf(lr[j] * inv);
  }
  __syncthreads();

  // 5. updates = attn @ v  (vT row-major per (b,e) is the B operand).
  // One A-fragment feeds both e-tiles (two accumulators).
  {
    int e0 = w * 32;
    const unsigned short* vb0 = vT + ((long)b * D_ + e0) * (long)N_;
    const unsigned short* vb1 = vb0 + (long)16 * N_;
    floatx8 c0 = zero8(), c1 = zero8();
    long lrow = (long)(lane & 15) * N_;
    for (int kk = 0; kk < N_ / 32; ++kk) {
      int ks = kk * 32;
      // prefetch streaming vT rows ~4KB ahead on both tile streams
      __builtin_prefetch(vb0 + lrow + ks + 2048, 0, 1);
      __builtin_prefetch(vb1 + lrow + ks + 2048, 0, 1);
      bf16x16 a = load_a<true>(attnp + ks, N_, lane, S_);
      c0 = wmma_bf(a, load_b(vb0 + ks, N_, lane), c0);
      c1 = wmma_bf(a, load_b(vb1 + ks, N_, lane), c1);
    }
    if (lane < 16) {
      float* up = updates + (long)b * S_ * D_;
#pragma unroll
      for (int r = 0; r < 8; ++r) {
        up[r * D_ + e0 + lane] = c0[r];
        up[r * D_ + e0 + 16 + lane] = c1[r];
      }
    }
  }
}

// Fused GRUCell + LayerNorm + MLP residual for 16 rows per block.
__global__ void gru_mlp_kernel(const float* __restrict__ updates,
                               float* __restrict__ slots,
                               const unsigned short* __restrict__ Wih,
                               const float* __restrict__ bih,
                               const unsigned short* __restrict__ Whh,
                               const float* __restrict__ bhh,
                               const unsigned short* __restrict__ W1,
                               const float* __restrict__ b1,
                               const unsigned short* __restrict__ W2,
                               const float* __restrict__ b2,
                               const float* __restrict__ g_mlp,
                               const float* __restrict__ be_mlp) {
  extern __shared__ char smem[];
  unsigned short* ub = (unsigned short*)smem;      // 16*256 bf16
  unsigned short* hb = ub + 16 * D_;               // 16*256 bf16
  float* hf = (float*)(hb + 16 * D_);              // 16*256 f32
  float* gx = hf + 16 * D_;                        // 16*768 f32
  float* gh = gx + 16 * 768;                       // 16*768 f32
  unsigned short* mb = (unsigned short*)(gh + 16 * 768); // 16*256 bf16
  unsigned short* tb = mb + 16 * D_;               // 16*256 bf16

  int w = threadIdx.x >> 5, lane = threadIdx.x & 31;
  long rowbase = (long)blockIdx.x * 16;

  for (int i = threadIdx.x; i < 16 * D_; i += 256) {
    float u = updates[rowbase * D_ + i];
    float h = slots[rowbase * D_ + i];
    ub[i] = f2bf(u);
    hb[i] = f2bf(h);
    hf[i] = h;
  }
  __syncthreads();

  // gx = u@W_ih^T + b_ih ; gh = h@W_hh^T + b_hh   (M=16, N=768, K=256)
  {
    bf16x16 au[8], ah[8];
#pragma unroll
    for (int kk = 0; kk < 8; ++kk) {
      au[kk] = load_a<false>(ub + kk * 32, D_, lane, 16);
      ah[kk] = load_a<false>(hb + kk * 32, D_, lane, 16);
    }
    for (int t = w; t < 48; t += 8) {
      int e0 = t * 16;
      floatx8 cx = zero8(), ch = zero8();
#pragma unroll
      for (int kk = 0; kk < 8; ++kk) {
        int ks = kk * 32;
        cx = wmma_bf(au[kk], load_b(Wih + e0 * D_ + ks, D_, lane), cx);
        ch = wmma_bf(ah[kk], load_b(Whh + e0 * D_ + ks, D_, lane), ch);
      }
      int e = e0 + (lane & 15);
      int rr = (lane & 16) ? 8 : 0;
      float bi = bih[e], bh2 = bhh[e];
#pragma unroll
      for (int r = 0; r < 8; ++r) {
        gx[(rr + r) * 768 + e] = cx[r] + bi;
        gh[(rr + r) * 768 + e] = ch[r] + bh2;
      }
    }
  }
  __syncthreads();

  // gates
  for (int i = threadIdx.x; i < 16 * D_; i += 256) {
    int row = i >> 8, d = i & (D_ - 1);
    float xr = gx[row * 768 + d];
    float xz = gx[row * 768 + D_ + d];
    float xg = gx[row * 768 + 2 * D_ + d];
    float hr = gh[row * 768 + d];
    float hz = gh[row * 768 + D_ + d];
    float hg = gh[row * 768 + 2 * D_ + d];
    float r = 1.f / (1.f + __expf(-(xr + hr)));
    float z = 1.f / (1.f + __expf(-(xz + hz)));
    float n = tanhf(xg + r * hg);
    hf[i] = (1.f - z) * n + z * hf[i];
  }
  __syncthreads();

  // LN(h') -> mb  (wave w handles rows w and w+8)
  for (int rsel = 0; rsel < 2; ++rsel) {
    int row = w + rsel * 8;
    float* hr2 = hf + row * D_;
    float v[8];
    float s = 0.f;
#pragma unroll
    for (int j = 0; j < 8; ++j) { v[j] = hr2[lane + 32 * j]; s += v[j]; }
    s = wsum(s);
    float mean = s * (1.f / D_);
    float q = 0.f;
#pragma unroll
    for (int j = 0; j < 8; ++j) { float d = v[j] - mean; q += d * d; }
    q = wsum(q);
    float rstd = rsqrtf(q * (1.f / D_) + EPS_);
#pragma unroll
    for (int j = 0; j < 8; ++j) {
      int idx = lane + 32 * j;
      mb[row * D_ + idx] =
          f2bf((v[j] - mean) * rstd * g_mlp[idx] + be_mlp[idx]);
    }
  }
  __syncthreads();

  // t = relu(m@W1^T + b1)  (shared A fragments, 2 tiles per wave)
  {
    bf16x16 am[8];
#pragma unroll
    for (int kk = 0; kk < 8; ++kk)
      am[kk] = load_a<false>(mb + kk * 32, D_, lane, 16);
    for (int tt = 0; tt < 2; ++tt) {
      int e0 = w * 32 + tt * 16;
      floatx8 c = zero8();
#pragma unroll
      for (int kk = 0; kk < 8; ++kk)
        c = wmma_bf(am[kk], load_b(W1 + e0 * D_ + kk * 32, D_, lane), c);
      int e = e0 + (lane & 15);
      int rr = (lane & 16) ? 8 : 0;
      float bias = b1[e];
#pragma unroll
      for (int r = 0; r < 8; ++r)
        tb[(rr + r) * D_ + e] = f2bf(fmaxf(c[r] + bias, 0.f));
    }
  }
  __syncthreads();

  // slots = h' + t@W2^T + b2
  {
    bf16x16 at[8];
#pragma unroll
    for (int kk = 0; kk < 8; ++kk)
      at[kk] = load_a<false>(tb + kk * 32, D_, lane, 16);
    for (int tt = 0; tt < 2; ++tt) {
      int e0 = w * 32 + tt * 16;
      floatx8 c = zero8();
#pragma unroll
      for (int kk = 0; kk < 8; ++kk)
        c = wmma_bf(at[kk], load_b(W2 + e0 * D_ + kk * 32, D_, lane), c);
      int e = e0 + (lane & 15);
      int rr = (lane & 16) ? 8 : 0;
      float bias = b2[e];
#pragma unroll
      for (int r = 0; r < 8; ++r)
        slots[(rowbase + rr + r) * D_ + e] =
            hf[(rr + r) * D_ + e] + c[r] + bias;
    }
  }
}

// ---------------- launch ----------------

extern "C" void kernel_launch(void* const* d_in, const int* in_sizes, int n_in,
                              void* d_out, int out_size, void* d_ws,
                              size_t ws_size, hipStream_t stream) {
  (void)in_sizes; (void)n_in; (void)out_size; (void)ws_size;
  const float* x          = (const float*)d_in[0];
  const float* noise      = (const float*)d_in[1];
  const float* slot_mu    = (const float*)d_in[2];
  const float* slot_sigma = (const float*)d_in[3];
  const float* Wq  = (const float*)d_in[4];
  const float* bq  = (const float*)d_in[5];
  const float* Wk  = (const float*)d_in[6];
  const float* bk  = (const float*)d_in[7];
  const float* Wv  = (const float*)d_in[8];
  const float* bv  = (const float*)d_in[9];
  const float* Wih = (const float*)d_in[10];
  const float* bih = (const float*)d_in[11];
  const float* Whh = (const float*)d_in[12];
  const float* bhh = (const float*)d_in[13];
  const float* W1  = (const float*)d_in[14];
  const float* b1  = (const float*)d_in[15];
  const float* W2  = (const float*)d_in[16];
  const float* b2  = (const float*)d_in[17];
  const float* g_in   = (const float*)d_in[18];
  const float* be_in  = (const float*)d_in[19];
  const float* g_sl   = (const float*)d_in[20];
  const float* be_sl  = (const float*)d_in[21];
  const float* g_mlp  = (const float*)d_in[22];
  const float* be_mlp = (const float*)d_in[23];

  char* ws = (char*)d_ws;
  size_t off = 0;
  auto alloc = [&](size_t bytes) -> void* {
    void* p = ws + off;
    off += (bytes + 255) & ~(size_t)255;
    return p;
  };
  unsigned short* xn   = (unsigned short*)alloc((size_t)B_ * N_ * D_ * 2);
  unsigned short* kbf  = (unsigned short*)alloc((size_t)B_ * N_ * D_ * 2);
  unsigned short* vT   = (unsigned short*)alloc((size_t)B_ * N_ * D_ * 2);
  float* slots         = (float*)alloc((size_t)B_ * S_ * D_ * 4);
  float* updates       = (float*)alloc((size_t)B_ * S_ * D_ * 4);
  unsigned short* Wq_b  = (unsigned short*)alloc((size_t)D_ * D_ * 2);
  unsigned short* Wk_b  = (unsigned short*)alloc((size_t)D_ * D_ * 2);
  unsigned short* Wv_b  = (unsigned short*)alloc((size_t)D_ * D_ * 2);
  unsigned short* Wih_b = (unsigned short*)alloc((size_t)3 * D_ * D_ * 2);
  unsigned short* Whh_b = (unsigned short*)alloc((size_t)3 * D_ * D_ * 2);
  unsigned short* W1_b  = (unsigned short*)alloc((size_t)D_ * D_ * 2);
  unsigned short* W2_b  = (unsigned short*)alloc((size_t)D_ * D_ * 2);

  const int nW = D_ * D_;        // 65536
  const int nWg = 3 * D_ * D_;   // 196608
  cast_bf16_kernel<<<nW / 256, 256, 0, stream>>>(Wq, Wq_b, nW);
  cast_bf16_kernel<<<nW / 256, 256, 0, stream>>>(Wk, Wk_b, nW);
  cast_bf16_kernel<<<nW / 256, 256, 0, stream>>>(Wv, Wv_b, nW);
  cast_bf16_kernel<<<nWg / 256, 256, 0, stream>>>(Wih, Wih_b, nWg);
  cast_bf16_kernel<<<nWg / 256, 256, 0, stream>>>(Whh, Whh_b, nWg);
  cast_bf16_kernel<<<nW / 256, 256, 0, stream>>>(W1, W1_b, nW);
  cast_bf16_kernel<<<nW / 256, 256, 0, stream>>>(W2, W2_b, nW);

  init_slots_kernel<<<(B_ * S_ * D_) / 256, 256, 0, stream>>>(
      noise, slot_mu, slot_sigma, slots);

  ln_x_kernel<<<(B_ * N_) / 8, 256, 0, stream>>>(x, g_in, be_in, xn);

  kv_kernel<<<(B_ * N_) / 32, 256, 0, stream>>>(xn, Wk_b, bk, Wv_b, bv, kbf,
                                                vT);

  const size_t smem_attn =
      (size_t)8 * N_ * 4 + 8 * N_ * 2 + 8 * D_ * 2 + 8 * D_ * 2;  // 204800 B
  const size_t smem_gru = (size_t)16 * D_ * 2 * 2 + 16 * D_ * 4 +
                          (size_t)16 * 768 * 4 * 2 +
                          (size_t)16 * D_ * 2 * 2;                // 147456 B

  for (int it = 0; it < ITERS_; ++it) {
    attn_kernel<<<B_, 256, smem_attn, stream>>>(slots, Wq_b, bq, kbf, vT, g_sl,
                                                be_sl, updates);
    gru_mlp_kernel<<<(B_ * S_) / 16, 256, smem_gru, stream>>>(
        updates, slots, Wih_b, bih, Whh_b, bhh, W1_b, b1, W2_b, b2, g_mlp,
        be_mlp);
  }

  hipMemcpyAsync(d_out, slots, (size_t)B_ * S_ * D_ * sizeof(float),
                 hipMemcpyDeviceToDevice, stream);
}
